// BERTTripleScorer_34737695490062
// MI455X (gfx1250) — matrix operand order repaired and checked by reading, not compiled, not run on hardware
//
#include <hip/hip_runtime.h>
#include <hip/hip_bf16.h>

typedef __bf16 bf16;
typedef __attribute__((ext_vector_type(16))) __bf16 v16bf;
typedef __attribute__((ext_vector_type(8)))  float  v8f;

#define D      768
#define NHEAD  12
#define FF     3072
#define L_ENTC 8192
#define NTOK   8704         /* L_ENT + R_REL */
#define NB     16384
#define MTAB   17536        /* 17473 valid rows padded to multiple of 64 */
#define ROW_TAIL0 8704
#define ROW_EDGE0 17408
#define ROW_CLS   17472

__device__ __forceinline__ float wave_sum(float x) {
#pragma unroll
  for (int off = 16; off > 0; off >>= 1) x += __shfl_xor(x, off, 32);
  return x;
}

/* ---- f32 (K x N, row-major) -> bf16 transposed (N x K, row-major)
   LDS-tiled so both global reads and writes are coalesced. grid=(N/32, K/32) ---- */
__global__ __launch_bounds__(256)
void transpose_to_bf16(const float* __restrict__ src, bf16* __restrict__ dst,
                       int K, int N) {
  __shared__ float tile[32][33];
  int n0 = blockIdx.x * 32;
  int k0 = blockIdx.y * 32;
  int tx = threadIdx.x & 31, ty = threadIdx.x >> 5;   /* ty in 0..7 */
#pragma unroll
  for (int r = 0; r < 4; ++r) {
    int k = k0 + ty + r * 8;
    tile[ty + r * 8][tx] = src[(long)k * N + n0 + tx];
  }
  __syncthreads();
#pragma unroll
  for (int r = 0; r < 4; ++r) {
    int n = n0 + ty + r * 8;
    dst[(long)n * K + k0 + tx] = (bf16)tile[tx][ty + r * 8];
  }
}

__global__ void concat_bias(const float* __restrict__ a, const float* __restrict__ b,
                            float* __restrict__ out) {
  int i = blockIdx.x * 256 + threadIdx.x;
  if (i < 768) out[i] = a[i];
  else if (i < 1536) out[i] = b[i - 768];
}

/* ---- distinct-token embedding + LN table (one wave per row) ---- */
__global__ __launch_bounds__(256)
void xtab_kernel(const float* __restrict__ tok, const float* __restrict__ rel,
                 const float* __restrict__ edge, const float* __restrict__ cls,
                 const float* __restrict__ seg, const float* __restrict__ pos,
                 const float* __restrict__ g, const float* __restrict__ bta,
                 bf16* __restrict__ Xtab, float* __restrict__ x0f) {
  int wave = threadIdx.x >> 5, lane = threadIdx.x & 31;
  int r = blockIdx.x * 8 + wave;
  if (r >= MTAB) return;
  bf16* out = Xtab + (long)r * D;
  const float* src = nullptr;
  int s = 0;
  if (r < ROW_TAIL0) { s = 1; src = (r < L_ENTC) ? tok + (long)r * D : rel + (long)(r - L_ENTC) * D; }
  else if (r < ROW_EDGE0) { int t = r - ROW_TAIL0; s = 3; src = (t < L_ENTC) ? tok + (long)t * D : rel + (long)(t - L_ENTC) * D; }
  else if (r < ROW_CLS) { s = 2; src = edge + (long)(r - ROW_EDGE0) * D; }
  else if (r == ROW_CLS) { s = 0; src = cls; }
  if (!src) {                       /* padding rows: zero them */
    for (int d = lane; d < D; d += 32) out[d] = (bf16)0.0f;
    return;
  }
  float vals[24], sum = 0.f, sq = 0.f;
#pragma unroll
  for (int i = 0; i < 24; ++i) {
    int d = lane + i * 32;
    float v = src[d] + seg[s * D + d] + pos[s * D + d];
    vals[i] = v; sum += v; sq += v * v;
  }
  sum = wave_sum(sum); sq = wave_sum(sq);
  float mean = sum * (1.0f / D);
  float var  = sq * (1.0f / D) - mean * mean;
  float rstd = rsqrtf(var + 1e-12f);
#pragma unroll
  for (int i = 0; i < 24; ++i) {
    int d = lane + i * 32;
    float y = (vals[i] - mean) * rstd * g[d] + bta[d];
    out[d] = (bf16)y;
    if (r == ROW_CLS) x0f[d] = y;
  }
}

/* ---- q0 = x0 @ Wq + bq (single shared row, f32) ---- */
__global__ void q0_kernel(const float* __restrict__ x0, const float* __restrict__ Wq,
                          const float* __restrict__ bq, float* __restrict__ q0) {
  int n = blockIdx.x * 256 + threadIdx.x;
  if (n >= D) return;
  float acc = bq[n];
  for (int k = 0; k < D; ++k) acc += x0[k] * Wq[(long)k * D + n];
  q0[n] = acc;
}

/* ---- bf16 WMMA GEMM: C[M,N] = A[M,K] @ Wt[N,K]^T + bias
   EPI: 0 = f32 store, 1 = bf16 store, 2 = gelu(tanh) -> bf16 store
   One wave computes a 64x64 tile via 4x4 grid of 16x16 wmma accumulators. ---- */
template <int EPI>
__global__ __launch_bounds__(256)
void gemm_bf16_kernel(const bf16* __restrict__ A, const bf16* __restrict__ Bt,
                      const float* __restrict__ bias, void* __restrict__ C,
                      int M, int N, int K) {
  int wave = threadIdx.x >> 5;
  int lane = threadIdx.x & 31;
  int ntiles = N >> 6;
  int g  = blockIdx.x * 8 + wave;
  int mt = g / ntiles, nt = g % ntiles;
  int m0 = mt * 64,    n0 = nt * 64;
  if (m0 >= M) return;
  int lhi = lane >> 4;       /* which 16-lane half */
  int llo = lane & 15;

  v8f acc[4][4];
#pragma unroll
  for (int j = 0; j < 4; ++j) {
    float bvv = bias ? bias[n0 + j * 16 + llo] : 0.0f;
    v8f z;
#pragma unroll
    for (int r = 0; r < 8; ++r) z[r] = bvv;
#pragma unroll
    for (int i = 0; i < 4; ++i) acc[i][j] = z;
  }

  /* A frag (16x32 bf16): lane half selects contiguous K 0-15 / 16-31 of A row m */
  const bf16* arow[4];
#pragma unroll
  for (int i = 0; i < 4; ++i) arow[i] = A + (long)(m0 + i * 16 + llo) * K + lhi * 16;
  /* B frag (32x16 bf16): lane half selects contiguous K 0-15 / 16-31 of Wt row n */
  const bf16* bcol[4];
#pragma unroll
  for (int j = 0; j < 4; ++j) bcol[j] = Bt + (long)(n0 + j * 16 + llo) * K + lhi * 16;

  for (int kk = 0; kk < K; kk += 32) {
    v16bf a[4], bfr[4];
#pragma unroll
    for (int i = 0; i < 4; ++i) {
      ((uint4*)&a[i])[0] = *(const uint4*)(arow[i] + kk);
      ((uint4*)&a[i])[1] = *(const uint4*)(arow[i] + kk + 8);
    }
#pragma unroll
    for (int j = 0; j < 4; ++j) {
      ((uint4*)&bfr[j])[0] = *(const uint4*)(bcol[j] + kk);
      ((uint4*)&bfr[j])[1] = *(const uint4*)(bcol[j] + kk + 8);
    }
#pragma unroll
    for (int i = 0; i < 4; ++i)
#pragma unroll
      for (int j = 0; j < 4; ++j)
        acc[i][j] = __builtin_amdgcn_wmma_f32_16x16x32_bf16(false, a[i], false, bfr[j],
                                                            (short)0, acc[i][j], false, false);
  }

#pragma unroll
  for (int i = 0; i < 4; ++i)
#pragma unroll
    for (int j = 0; j < 4; ++j) {
      int col = n0 + j * 16 + llo;
#pragma unroll
      for (int r = 0; r < 8; ++r) {
        int row = m0 + i * 16 + r + 8 * lhi;
        float v = acc[i][j][r];
        if (EPI == 0) {
          ((float*)C)[(long)row * N + col] = v;
        } else if (EPI == 1) {
          ((bf16*)C)[(long)row * N + col] = (bf16)v;
        } else {
          float t  = 0.7978845608028654f * (v + 0.044715f * v * v * v);
          float gv = 0.5f * v * (1.0f + tanhf(t));
          ((bf16*)C)[(long)row * N + col] = (bf16)gv;
        }
      }
    }
}

/* ---- attention: one wave per (triple, head); gathers K/V table rows ---- */
__global__ __launch_bounds__(256)
void attn_kernel(const int* __restrict__ head_ids, const int* __restrict__ et_ids,
                 const int* __restrict__ tail_ids, const float* __restrict__ q0,
                 const bf16* __restrict__ KVtab, bf16* __restrict__ ctx) {
  int wave = threadIdx.x >> 5, lane = threadIdx.x & 31;
  int gidx = blockIdx.x * 8 + wave;
  int b = gidx / NHEAD, h = gidx % NHEAD;
  if (b >= NB) return;
  int rows[4];
  rows[0] = ROW_CLS;
  rows[1] = head_ids[b];
  rows[2] = ROW_EDGE0 + et_ids[b];
  rows[3] = ROW_TAIL0 + tail_ids[b];
  float qa = q0[h * 64 + lane], qb = q0[h * 64 + 32 + lane];
  float s[4];
#pragma unroll
  for (int j = 0; j < 4; ++j) {
    const bf16* kr = KVtab + (long)rows[j] * 1536 + h * 64;
    float p = qa * (float)kr[lane] + qb * (float)kr[lane + 32];
    s[j] = wave_sum(p) * 0.125f;
  }
  float mx = fmaxf(fmaxf(s[0], s[1]), fmaxf(s[2], s[3]));
  float e[4], sum = 0.f;
#pragma unroll
  for (int j = 0; j < 4; ++j) { e[j] = __expf(s[j] - mx); sum += e[j]; }
  float inv = 1.0f / sum;
  float ca = 0.f, cb = 0.f;
#pragma unroll
  for (int j = 0; j < 4; ++j) {
    const bf16* vr = KVtab + (long)rows[j] * 1536 + 768 + h * 64;
    float p = e[j] * inv;
    ca += p * (float)vr[lane];
    cb += p * (float)vr[lane + 32];
  }
  ctx[(long)b * D + h * 64 + lane]      = (bf16)ca;
  ctx[(long)b * D + h * 64 + 32 + lane] = (bf16)cb;
}

/* ---- LN1: x1 = LN(x0 + attnWoOut) ; writes f32 (residual) + bf16 (gemm in) ---- */
__global__ __launch_bounds__(256)
void ln1_kernel(const float* __restrict__ wo, const float* __restrict__ x0,
                const float* __restrict__ g, const float* __restrict__ bta,
                float* __restrict__ x1f, bf16* __restrict__ x1b) {
  int wave = threadIdx.x >> 5, lane = threadIdx.x & 31;
  int r = blockIdx.x * 8 + wave;
  if (r >= NB) return;
  const float* in = wo + (long)r * D;
  float vals[24], sum = 0.f, sq = 0.f;
#pragma unroll
  for (int i = 0; i < 24; ++i) {
    int d = lane + i * 32;
    float v = in[d] + x0[d];
    vals[i] = v; sum += v; sq += v * v;
  }
  sum = wave_sum(sum); sq = wave_sum(sq);
  float mean = sum * (1.0f / D);
  float rstd = rsqrtf(sq * (1.0f / D) - mean * mean + 1e-12f);
#pragma unroll
  for (int i = 0; i < 24; ++i) {
    int d = lane + i * 32;
    float y = (vals[i] - mean) * rstd * g[d] + bta[d];
    x1f[(long)r * D + d] = y;
    x1b[(long)r * D + d] = (bf16)y;
  }
}

/* ---- LN2 + score: scores[b] = LN(x1 + ffn2) . head_w + head_b ---- */
__global__ __launch_bounds__(256)
void ln2_score_kernel(const float* __restrict__ f2, const float* __restrict__ x1f,
                      const float* __restrict__ g, const float* __restrict__ bta,
                      const float* __restrict__ hw, const float* __restrict__ hb,
                      float* __restrict__ scores) {
  int wave = threadIdx.x >> 5, lane = threadIdx.x & 31;
  int r = blockIdx.x * 8 + wave;
  if (r >= NB) return;
  float vals[24], sum = 0.f, sq = 0.f;
#pragma unroll
  for (int i = 0; i < 24; ++i) {
    int d = lane + i * 32;
    float v = f2[(long)r * D + d] + x1f[(long)r * D + d];
    vals[i] = v; sum += v; sq += v * v;
  }
  sum = wave_sum(sum); sq = wave_sum(sq);
  float mean = sum * (1.0f / D);
  float rstd = rsqrtf(sq * (1.0f / D) - mean * mean + 1e-12f);
  float acc = 0.f;
#pragma unroll
  for (int i = 0; i < 24; ++i) {
    int d = lane + i * 32;
    float y = (vals[i] - mean) * rstd * g[d] + bta[d];
    acc += y * hw[d];
  }
  acc = wave_sum(acc);
  if (lane == 0) scores[r] = acc + hb[0];
}

static inline int gemm_blocks(int M, int N) {
  int waves = (M / 64) * (N / 64);
  return (waves + 7) / 8;
}

extern "C" void kernel_launch(void* const* d_in, const int* in_sizes, int n_in,
                              void* d_out, int out_size, void* d_ws, size_t ws_size,
                              hipStream_t stream) {
  (void)in_sizes; (void)n_in; (void)out_size; (void)ws_size;
  const int*   head_ids = (const int*)d_in[0];
  const int*   et_ids   = (const int*)d_in[1];
  const int*   tail_ids = (const int*)d_in[2];
  const float* tok  = (const float*)d_in[4];
  const float* rel  = (const float*)d_in[5];
  const float* edge = (const float*)d_in[6];
  const float* cls  = (const float*)d_in[7];
  const float* seg  = (const float*)d_in[8];
  const float* pos  = (const float*)d_in[9];
  const float* lng  = (const float*)d_in[10];
  const float* lnb  = (const float*)d_in[11];
  const float* Wq   = (const float*)d_in[12];
  const float* bq   = (const float*)d_in[13];
  const float* Wk   = (const float*)d_in[14];
  const float* bk   = (const float*)d_in[15];
  const float* Wv   = (const float*)d_in[16];
  const float* bv   = (const float*)d_in[17];
  const float* Wo   = (const float*)d_in[18];
  const float* bo   = (const float*)d_in[19];
  const float* ln1g = (const float*)d_in[20];
  const float* ln1b = (const float*)d_in[21];
  const float* W1   = (const float*)d_in[22];
  const float* b1   = (const float*)d_in[23];
  const float* W2   = (const float*)d_in[24];
  const float* b2   = (const float*)d_in[25];
  const float* ln2g = (const float*)d_in[26];
  const float* ln2b = (const float*)d_in[27];
  const float* hw   = (const float*)d_in[28];
  const float* hb   = (const float*)d_in[29];

  char* wsb = (char*)d_ws;
  size_t off = 0;
  auto alloc = [&](size_t bytes) -> char* {
    char* p = wsb + off;
    off += (bytes + 255) & ~(size_t)255;
    return p;
  };
  bf16*  Wt_kv  = (bf16*)alloc((size_t)1536 * 768 * 2);
  bf16*  Wt_o   = (bf16*)alloc((size_t)768 * 768 * 2);
  bf16*  Wt_1   = (bf16*)alloc((size_t)3072 * 768 * 2);
  bf16*  Wt_2   = (bf16*)alloc((size_t)768 * 3072 * 2);
  float* biaskv = (float*)alloc(1536 * 4);
  float* x0f    = (float*)alloc(768 * 4);
  float* q0     = (float*)alloc(768 * 4);
  bf16*  Xtab   = (bf16*)alloc((size_t)MTAB * 768 * 2);
  bf16*  KVtab  = (bf16*)alloc((size_t)MTAB * 1536 * 2);
  bf16*  ctx    = (bf16*)alloc((size_t)NB * 768 * 2);
  float* tmpf   = (float*)alloc((size_t)NB * 768 * 4);   /* Wo-out, later FFN2-out */
  float* x1f    = (float*)alloc((size_t)NB * 768 * 4);
  bf16*  x1b    = (bf16*)alloc((size_t)NB * 768 * 2);
  bf16*  hgelu  = (bf16*)alloc((size_t)NB * 3072 * 2);

  /* weight prep: bf16 transposed copies (coalesced LDS tile transpose) */
  transpose_to_bf16<<<dim3(768 / 32, 768 / 32), 256, 0, stream>>>(Wk, Wt_kv, 768, 768);
  transpose_to_bf16<<<dim3(768 / 32, 768 / 32), 256, 0, stream>>>(Wv, Wt_kv + (size_t)768 * 768, 768, 768);
  transpose_to_bf16<<<dim3(768 / 32, 768 / 32), 256, 0, stream>>>(Wo, Wt_o, 768, 768);
  transpose_to_bf16<<<dim3(3072 / 32, 768 / 32), 256, 0, stream>>>(W1, Wt_1, 768, 3072);
  transpose_to_bf16<<<dim3(768 / 32, 3072 / 32), 256, 0, stream>>>(W2, Wt_2, 3072, 768);
  concat_bias<<<6, 256, 0, stream>>>(bk, bv, biaskv);

  /* distinct-token embed + LN table (also emits shared CLS row x0) */
  xtab_kernel<<<(MTAB + 7) / 8, 256, 0, stream>>>(tok, rel, edge, cls, seg, pos,
                                                  lng, lnb, Xtab, x0f);
  q0_kernel<<<3, 256, 0, stream>>>(x0f, Wq, bq, q0);

  /* KV table GEMM: [17536 x 768] @ [768 x 1536] -> bf16 */
  gemm_bf16_kernel<1><<<gemm_blocks(MTAB, 1536), 256, 0, stream>>>(
      Xtab, Wt_kv, biaskv, KVtab, MTAB, 1536, 768);

  /* attention (CLS query only), gathers from L2-resident KV table */
  attn_kernel<<<(NB * NHEAD + 7) / 8, 256, 0, stream>>>(head_ids, et_ids, tail_ids,
                                                        q0, KVtab, ctx);

  /* Wo projection for CLS rows */
  gemm_bf16_kernel<0><<<gemm_blocks(NB, 768), 256, 0, stream>>>(
      ctx, Wt_o, bo, tmpf, NB, 768, 768);
  ln1_kernel<<<NB / 8, 256, 0, stream>>>(tmpf, x0f, ln1g, ln1b, x1f, x1b);

  /* FFN */
  gemm_bf16_kernel<2><<<gemm_blocks(NB, 3072), 256, 0, stream>>>(
      x1b, Wt_1, b1, hgelu, NB, 3072, 768);
  gemm_bf16_kernel<0><<<gemm_blocks(NB, 768), 256, 0, stream>>>(
      hgelu, Wt_2, b2, tmpf, NB, 768, 3072);

  ln2_score_kernel<<<NB / 8, 256, 0, stream>>>(tmpf, x1f, ln2g, ln2b, hw, hb,
                                               (float*)d_out);
}
